// maskselfattention_90890097918339
// MI455X (gfx1250) — matrix operand order repaired and checked by reading, not compiled, not run on hardware
//
#include <hip/hip_runtime.h>
#include <stdint.h>

#define B_  2
#define S_  2048
#define D_  1024
#define H_  16
#define HD_ 64

typedef __attribute__((ext_vector_type(16))) __bf16 v16bf;
typedef __attribute__((ext_vector_type(8)))  __bf16 v8bf;
typedef __attribute__((ext_vector_type(8)))  float  v8f;
typedef int v4i __attribute__((vector_size(16)));   // matches builtin param type

#define AS1 __attribute__((address_space(1)))
#define AS3 __attribute__((address_space(3)))

#if __has_builtin(__builtin_amdgcn_global_load_async_to_lds_b128)
#define ASYNC_LDS 1
#endif

// native fp32 -> bf16 (RNE); backend picks the hardware convert
__device__ __forceinline__ __bf16 f2bf(float f) { return (__bf16)f; }

// Build a v16bf fragment from two 16-byte (8 x bf16) aligned chunks.
__device__ __forceinline__ v16bf make_frag(const __bf16* lo, const __bf16* hi) {
    union { v16bf v; v8bf h[2]; } r;
    r.h[0] = *(const v8bf*)lo;
    r.h[1] = *(const v8bf*)hi;
    return r.v;
}

__device__ __forceinline__ v8f wmma_bf16(v16bf a, v16bf b, v8f c) {
    return __builtin_amdgcn_wmma_f32_16x16x32_bf16(
        /*neg_a=*/false, a, /*neg_b=*/false, b,
        /*c_mod=*/(short)0, c, /*reuse_a=*/false, /*reuse_b=*/false);
}

#ifdef ASYNC_LDS
// (global v4i*, lds v4i*, imm offset, imm cpol); LDS offset = low 32 bits of
// the generic address.
__device__ __forceinline__ void async_cp16(const void* gsrc, void* ldst) {
    __builtin_amdgcn_global_load_async_to_lds_b128(
        (AS1 v4i*)(uintptr_t)gsrc,
        (AS3 v4i*)(uint32_t)(uintptr_t)ldst,
        0, 0);
}
__device__ __forceinline__ void wait_async0() {
#if __has_builtin(__builtin_amdgcn_s_wait_asynccnt)
    __builtin_amdgcn_s_wait_asynccnt(0);
#else
    asm volatile("s_wait_asynccnt 0" ::: "memory");
#endif
}
#endif

// ---------------------------------------------------------------------------
// Kernel 1: fused QKV projection.  O = bf16(x) @ bf16(W), fp32 accumulate,
// output stored bf16 in [B, H, S, hd] layout.  128x64 block tile, BK = 32;
// each of the 8 waves owns a 32x32 accumulator (4 WMMAs / K-step).
// ---------------------------------------------------------------------------
__global__ __launch_bounds__(256) void qkv_gemm(
    const float* __restrict__ x,
    const float* __restrict__ Wq, const float* __restrict__ Wk,
    const float* __restrict__ Wv,
    __bf16* __restrict__ Qo, __bf16* __restrict__ Ko, __bf16* __restrict__ Vo)
{
    const int tid  = threadIdx.x;
    const int lane = tid & 31;
    const int wave = tid >> 5;
    const int m0   = blockIdx.y * 128;  // row tile over B*S
    const int n0   = blockIdx.x * 64;   // col tile over D
    const float* W = (blockIdx.z == 0) ? Wq : (blockIdx.z == 1) ? Wk : Wv;
    __bf16*      O = (blockIdx.z == 0) ? Qo : (blockIdx.z == 1) ? Ko : Vo;

    const int LD = 48;                  // padded stride (96 B, 16B-aligned)
    __shared__ __align__(16) __bf16 xs[128 * LD];  // x tile, row-major
    __shared__ __align__(16) __bf16 wt[64 * LD];   // W tile, TRANSPOSED: wt[n][k]

    v8f acc00 = {}, acc01 = {}, acc10 = {}, acc11 = {};

    const int wrow = (wave & 3) * 32;   // 4 row groups of 32
    const int wcol = (wave >> 2) * 32;  // 2 col groups of 32

    for (int k0 = 0; k0 < D_; k0 += 32) {
        __syncthreads();
        {   // x tile: 128 rows x 32 cols fp32 -> bf16 (16 elements / thread)
            int r = tid >> 1;
            int c = (tid & 1) * 16;
            const float* src = x + (size_t)(m0 + r) * D_ + k0 + c;
            float4 f0 = *(const float4*)src;
            float4 f1 = *(const float4*)(src + 4);
            float4 f2 = *(const float4*)(src + 8);
            float4 f3 = *(const float4*)(src + 12);
            union { v8bf v; uint4 u; } p0, p1;
            p0.v[0] = f2bf(f0.x); p0.v[1] = f2bf(f0.y);
            p0.v[2] = f2bf(f0.z); p0.v[3] = f2bf(f0.w);
            p0.v[4] = f2bf(f1.x); p0.v[5] = f2bf(f1.y);
            p0.v[6] = f2bf(f1.z); p0.v[7] = f2bf(f1.w);
            p1.v[0] = f2bf(f2.x); p1.v[1] = f2bf(f2.y);
            p1.v[2] = f2bf(f2.z); p1.v[3] = f2bf(f2.w);
            p1.v[4] = f2bf(f3.x); p1.v[5] = f2bf(f3.y);
            p1.v[6] = f2bf(f3.z); p1.v[7] = f2bf(f3.w);
            *(uint4*)&xs[r * LD + c]     = p0.u;
            *(uint4*)&xs[r * LD + c + 8] = p1.u;
        }
        {   // W tile: 32 k-rows x 64 n-cols, store transposed wt[n][k]
            int kk = tid >> 3;
            int nn = (tid & 7) * 8;
            const float* src = W + (size_t)(k0 + kk) * D_ + n0 + nn;
            float4 f0 = *(const float4*)src;
            float4 f1 = *(const float4*)(src + 4);
            float f[8] = {f0.x, f0.y, f0.z, f0.w, f1.x, f1.y, f1.z, f1.w};
            #pragma unroll
            for (int j = 0; j < 8; ++j)
                wt[(nn + j) * LD + kk] = f2bf(f[j]);
        }
        __syncthreads();

        // A fragments (16x32): lane&15 = row; hi half shifts K by 8 per 16-K half
        int akoff = (lane >> 4) * 8;
        int ar0   = (wrow + (lane & 15)) * LD;
        int ar1   = (wrow + 16 + (lane & 15)) * LD;
        v16bf a0 = make_frag(&xs[ar0 + akoff], &xs[ar0 + 16 + akoff]);
        v16bf a1 = make_frag(&xs[ar1 + akoff], &xs[ar1 + 16 + akoff]);
        // B fragments (32x16): lane&15 = col n (row of wt); hi half shifts K by 8 per 16-K half
        int bk  = (lane >> 4) * 8;
        int bn0 = (wcol + (lane & 15)) * LD;
        int bn1 = (wcol + 16 + (lane & 15)) * LD;
        v16bf b0 = make_frag(&wt[bn0 + bk], &wt[bn0 + 16 + bk]);
        v16bf b1 = make_frag(&wt[bn1 + bk], &wt[bn1 + 16 + bk]);
        acc00 = wmma_bf16(a0, b0, acc00);
        acc01 = wmma_bf16(a0, b1, acc01);
        acc10 = wmma_bf16(a1, b0, acc10);
        acc11 = wmma_bf16(a1, b1, acc11);
    }

    // Scatter out as bf16 in [B, H, S, hd]
    #pragma unroll
    for (int rt = 0; rt < 2; ++rt) {
        #pragma unroll
        for (int t = 0; t < 2; ++t) {
            v8f acc = rt ? (t ? acc11 : acc10) : (t ? acc01 : acc00);
            int ncol = n0 + wcol + t * 16 + (lane & 15);
            int h = ncol >> 6, dh = ncol & 63;
            #pragma unroll
            for (int r = 0; r < 8; ++r) {
                int m = m0 + wrow + rt * 16 + r + ((lane >> 4) << 3);
                int b = m >> 11, s = m & (S_ - 1);
                O[(((size_t)b * H_ + h) * S_ + s) * HD_ + dh] = f2bf(acc[r]);
            }
        }
    }
}

// ---------------------------------------------------------------------------
// Kernel 2: causal flash attention, online softmax, bf16 WMMA / fp32 stats.
// 4 waves per block; wave w owns 16 query rows of a 64-row q tile.
// Keys processed in chunks of 32 (only chunks under the causal diagonal).
// K tile copied global->LDS with async-LDS DMA when available.
// ---------------------------------------------------------------------------
__global__ __launch_bounds__(128) void flash_attn(
    const __bf16* __restrict__ Q, const __bf16* __restrict__ K,
    const __bf16* __restrict__ V, float* __restrict__ Out)
{
    const int tid   = threadIdx.x;
    const int lane  = tid & 31;
    const int wave  = tid >> 5;
    const int qtile = blockIdx.x;
    const int b = blockIdx.y >> 4, h = blockIdx.y & 15;

    const size_t head_off = ((size_t)b * H_ + h) * S_ * HD_;
    const __bf16* Qh = Q + head_off;
    const __bf16* Kh = K + head_off;
    const __bf16* Vh = V + head_off;

    const int KS_LD = 72, VT_LD = 40, PS_LD = 40;   // all 16B-aligned strides
    __shared__ __align__(16) __bf16 ks[32 * KS_LD];      // K chunk, row-major [key][d]
    __shared__ __align__(16) __bf16 vt[64 * VT_LD];      // V chunk, transposed [d][key]
    __shared__ __align__(16) __bf16 ps[4][16 * PS_LD];   // per-wave P transpose staging

    const int qbase    = qtile * 64 + wave * 16;
    const int hi       = lane >> 4;        // 0 or 1
    const int qrow_off = hi << 3;          // C-layout: hi half holds rows +8
    const int col      = lane & 15;

    // Q A-fragments for this wave (hd split 0..31 / 32..63)
    const int  arow  = qbase + col;
    const int  akoff = hi * 8;
    const __bf16* qp = Qh + (size_t)arow * HD_;
    v16bf aq0 = make_frag(qp + akoff,      qp + 16 + akoff);
    v16bf aq1 = make_frag(qp + 32 + akoff, qp + 48 + akoff);

    v8f o0 = {}, o1 = {}, o2 = {}, o3 = {};
    float m8[8], l8[8];
    #pragma unroll
    for (int r = 0; r < 8; ++r) { m8[r] = -__builtin_inff(); l8[r] = 0.f; }

    const int nchunks = (qtile + 1) * 2;   // keys 0 .. qtile*64+63
    for (int c = 0; c < nchunks; ++c) {
        const int kbase = c * 32;
        __syncthreads();
        {   // K chunk 32x64 -> ks (row-major); async DMA when available
            int r  = tid >> 2;
            int cc = (tid & 3) * 16;
            const __bf16* src = Kh + (size_t)(kbase + r) * HD_ + cc;
            __bf16* dst = &ks[r * KS_LD + cc];
#ifdef ASYNC_LDS
            async_cp16(src,     dst);
            async_cp16(src + 8, dst + 8);
#else
            *(uint4*)dst       = *(const uint4*)src;
            *(uint4*)(dst + 8) = *(const uint4*)(src + 8);
#endif
        }
        {   // V chunk 32x64 -> vt (transposed)
            int r  = tid >> 2;
            int cc = (tid & 3) * 16;
            const __bf16* src = Vh + (size_t)(kbase + r) * HD_ + cc;
            #pragma unroll
            for (int j = 0; j < 16; ++j)
                vt[(cc + j) * VT_LD + r] = src[j];
        }
        // speculative prefetch of the next chunk (stays inside the Q/K/V ws)
        if (c + 1 < nchunks) {
            __builtin_prefetch(Kh + (size_t)(kbase + 32 + (tid >> 2)) * HD_, 0, 1);
            __builtin_prefetch(Vh + (size_t)(kbase + 32 + (tid >> 2)) * HD_, 0, 1);
        }
#ifdef ASYNC_LDS
        wait_async0();              // s_barrier does not cover ASYNCcnt
#endif
        __syncthreads();

        // scores S = Q . K^T : 16 x 32 as two 16x16 tiles, K-depth 64 = 2 WMMAs each
        v8f s0 = {}, s1 = {};
        const int bkoff = hi * 8;
        {
            const __bf16* kr = &ks[col * KS_LD];           // keys kbase+0..15
            s0 = wmma_bf16(aq0, make_frag(kr + bkoff,      kr + 16 + bkoff),      s0);
            s0 = wmma_bf16(aq1, make_frag(kr + 32 + bkoff, kr + 48 + bkoff), s0);
        }
        {
            const __bf16* kr = &ks[(16 + col) * KS_LD];    // keys kbase+16..31
            s1 = wmma_bf16(aq0, make_frag(kr + bkoff,      kr + 16 + bkoff),      s1);
            s1 = wmma_bf16(aq1, make_frag(kr + 32 + bkoff, kr + 48 + bkoff), s1);
        }

        // scale + causal mask + online softmax
        const int kcol = kbase + col;
        #pragma unroll
        for (int r = 0; r < 8; ++r) {
            int qg = qbase + qrow_off + r;
            float x0 = s0[r] * 0.125f;                     // 1/sqrt(64)
            float x1 = s1[r] * 0.125f;
            if (kcol > qg)      x0 = -__builtin_inff();
            if (kcol + 16 > qg) x1 = -__builtin_inff();

            float v = fmaxf(x0, x1);                       // row max across 16 lanes/half
            v = fmaxf(v, __shfl_xor(v, 1));
            v = fmaxf(v, __shfl_xor(v, 2));
            v = fmaxf(v, __shfl_xor(v, 4));
            v = fmaxf(v, __shfl_xor(v, 8));
            float mnew = fmaxf(m8[r], v);
            float corr = __expf(m8[r] - mnew);
            float p0   = __expf(x0 - mnew);
            float p1   = __expf(x1 - mnew);
            float rs   = p0 + p1;
            rs += __shfl_xor(rs, 1);
            rs += __shfl_xor(rs, 2);
            rs += __shfl_xor(rs, 4);
            rs += __shfl_xor(rs, 8);
            l8[r] = l8[r] * corr + rs;
            m8[r] = mnew;
            o0[r] *= corr; o1[r] *= corr; o2[r] *= corr; o3[r] *= corr;

            // stage P (C-layout) -> LDS for re-read in A-layout
            int prow = qrow_off + r;
            ps[wave][prow * PS_LD + col]      = f2bf(p0);
            ps[wave][prow * PS_LD + 16 + col] = f2bf(p1);
        }

        // same-wave LDS ops are in-order; wait for the staging stores anyway
        asm volatile("s_wait_dscnt 0" ::: "memory");

        // P (16x32) A-fragment, then O += P . V  (4 hd tiles, one WMMA each)
        const __bf16* pp = &ps[wave][col * PS_LD];
        v16bf ap = make_frag(pp + akoff, pp + 16 + akoff);
        o0 = wmma_bf16(ap, make_frag(&vt[(0*16 + col) * VT_LD + bkoff],
                                     &vt[(0*16 + col) * VT_LD + 16 + bkoff]), o0);
        o1 = wmma_bf16(ap, make_frag(&vt[(1*16 + col) * VT_LD + bkoff],
                                     &vt[(1*16 + col) * VT_LD + 16 + bkoff]), o1);
        o2 = wmma_bf16(ap, make_frag(&vt[(2*16 + col) * VT_LD + bkoff],
                                     &vt[(2*16 + col) * VT_LD + 16 + bkoff]), o2);
        o3 = wmma_bf16(ap, make_frag(&vt[(3*16 + col) * VT_LD + bkoff],
                                     &vt[(3*16 + col) * VT_LD + 16 + bkoff]), o3);
    }

    // epilogue: normalize and store fp32 out[b, s, h*64 + dh]
    #pragma unroll
    for (int r = 0; r < 8; ++r) {
        int s = qbase + qrow_off + r;
        float inv = 1.0f / l8[r];
        float* orow = Out + ((size_t)b * S_ + s) * D_ + h * HD_;
        orow[ 0 + col] = o0[r] * inv;
        orow[16 + col] = o1[r] * inv;
        orow[32 + col] = o2[r] * inv;
        orow[48 + col] = o3[r] * inv;
    }
}

extern "C" void kernel_launch(void* const* d_in, const int* in_sizes, int n_in,
                              void* d_out, int out_size, void* d_ws, size_t ws_size,
                              hipStream_t stream) {
    (void)in_sizes; (void)n_in; (void)out_size; (void)ws_size;
    const float* x  = (const float*)d_in[0];
    const float* Wq = (const float*)d_in[1];
    const float* Wk = (const float*)d_in[2];
    const float* Wv = (const float*)d_in[3];
    float* out = (float*)d_out;

    const size_t per = (size_t)B_ * H_ * S_ * HD_;   // 4M elements each
    __bf16* qb = (__bf16*)d_ws;
    __bf16* kb = qb + per;
    __bf16* vb = kb + per;

    dim3 g1(D_ / 64, (B_ * S_) / 128, 3);            // 16 x 32 x 3 blocks
    qkv_gemm<<<g1, 256, 0, stream>>>(x, Wq, Wk, Wv, qb, kb, vb);

    dim3 g2(S_ / 64, B_ * H_);                       // 32 x 32 blocks
    flash_attn<<<g2, 128, 0, stream>>>(qb, kb, vb, out);
}
